// PortillaSimoncelli_72275709657321
// MI455X (gfx1250) — compile-verified, matrix operand order checked by reading, not run on hardware
//
#include <hip/hip_runtime.h>
#include <hip/hip_bf16.h>
#include <math.h>

#define PI_F 3.14159265358979f
#define RB 256          // reduction partial blocks
#define GW 512          // gram waves
#define FLT_BIG 3.402823466e38f

typedef float    v2f  __attribute__((ext_vector_type(2)));
typedef float    v8f  __attribute__((ext_vector_type(8)));
typedef _Float16 v16h __attribute__((ext_vector_type(16)));

// ---------------------------------------------------------------- helpers
__device__ __forceinline__ void ps_polar(int k, int l, int n, int lev,
                                         float& lr, float& ang) {
    int h = n >> 1;
    int i = k ^ h, j = l ^ h;              // fftshifted coords (n even, pow2)
    float y = (float)(i - h) / (float)h;
    float x = (float)(j - h) / (float)h;
    if (i == h && j == h) { x = -1.0f / (float)h; y = 0.0f; } // center fix
    ang = atan2f(y, x);
    lr  = 0.5f * log2f(x * x + y * y) - (float)lev;
}
__device__ __forceinline__ float ps_hi(float lr, float shift) {
    float d = fminf(fmaxf(lr - shift, -1.0f), 0.0f);
    return cosf(1.57079632679f * d);
}
__device__ __forceinline__ float ps_lo(float lr, float shift) {
    float h = ps_hi(lr, shift);
    float t = fminf(fmaxf(1.0f - h * h, 0.0f), 1.0f);
    return sqrtf(t);
}

// ---------------------------------------------------------------- FFT (LDS)
__global__ void k_fft_lines(float2* data, int n, int logn, int stride,
                            int lstride, int inverse) {
    __shared__ float2 sm[1024];
    float2* base = data + (size_t)blockIdx.x * (size_t)lstride;
    for (int i = threadIdx.x; i < n; i += blockDim.x) {
        int r = (int)(__brev((unsigned)i) >> (32 - logn));
        sm[r] = base[(size_t)i * stride];
    }
    __syncthreads();
    for (int s = 1; s <= logn; ++s) {
        int step = 1 << s, half = step >> 1;
        float sgn = inverse ? 2.0f : -2.0f;
        for (int t = threadIdx.x; t < (n >> 1); t += blockDim.x) {
            int g = t / half, j = t - g * half;
            int idx = g * step + j;
            float ang = sgn * PI_F * (float)j / (float)step;
            float c = cosf(ang), si = sinf(ang);
            float2 u = sm[idx], v = sm[idx + half];
            float wr = v.x * c - v.y * si;
            float wi = v.x * si + v.y * c;
            sm[idx]        = make_float2(u.x + wr, u.y + wi);
            sm[idx + half] = make_float2(u.x - wr, u.y - wi);
        }
        __syncthreads();
    }
    float scale = inverse ? 1.0f / (float)n : 1.0f;
    for (int i = threadIdx.x; i < n; i += blockDim.x) {
        float2 v = sm[i];
        base[(size_t)i * stride] = make_float2(v.x * scale, v.y * scale);
    }
}

// ---------------------------------------------------------------- pointwise
__global__ void k_memset_f32(float* p, float v, size_t n) {
    for (size_t i = blockIdx.x * (size_t)blockDim.x + threadIdx.x; i < n;
         i += (size_t)gridDim.x * blockDim.x) p[i] = v;
}
__global__ void k_r2c(const float* s, float2* d, size_t n, const float* meanp) {
    float m = meanp ? *meanp : 0.0f;
    for (size_t i = blockIdx.x * (size_t)blockDim.x + threadIdx.x; i < n;
         i += (size_t)gridDim.x * blockDim.x) d[i] = make_float2(s[i] - m, 0.0f);
}
__global__ void k_c2real(const float2* s, float* d, size_t n) {
    for (size_t i = blockIdx.x * (size_t)blockDim.x + threadIdx.x; i < n;
         i += (size_t)gridDim.x * blockDim.x) d[i] = s[i].x;
}
__global__ void k_power(float2* z, size_t n) {
    for (size_t i = blockIdx.x * (size_t)blockDim.x + threadIdx.x; i < n;
         i += (size_t)gridDim.x * blockDim.x) {
        float2 v = z[i];
        z[i] = make_float2(v.x * v.x + v.y * v.y, 0.0f);
    }
}
__global__ void k_cscale(float2* z, size_t n, float s) {
    for (size_t i = blockIdx.x * (size_t)blockDim.x + threadIdx.x; i < n;
         i += (size_t)gridDim.x * blockDim.x) {
        float2 v = z[i];
        z[i] = make_float2(v.x * s, v.y * s);
    }
}
__global__ void k_center(float* x, size_t n, const float* meanp) {
    float m = *meanp;
    for (size_t i = blockIdx.x * (size_t)blockDim.x + threadIdx.x; i < n;
         i += (size_t)gridDim.x * blockDim.x) x[i] -= m;
}
__global__ void k_band_to_planes(const float2* z, size_t n, float* mag, float* re) {
    for (size_t i = blockIdx.x * (size_t)blockDim.x + threadIdx.x; i < n;
         i += (size_t)gridDim.x * blockDim.x) {
        float2 v = z[i];
        mag[i] = sqrtf(v.x * v.x + v.y * v.y);
        re[i]  = v.x;
    }
}
__global__ void k_dp_planes(const float2* u, size_t n, float* pm, float* dre,
                            float* dim) {
    for (size_t i = blockIdx.x * (size_t)blockDim.x + threadIdx.x; i < n;
         i += (size_t)gridDim.x * blockDim.x) {
        float2 v = u[i];
        float m = sqrtf(v.x * v.x + v.y * v.y);
        float inv = 1.0f / (m + 1e-12f);
        pm[i]  = m;
        dre[i] = (v.x * v.x - v.y * v.y) * inv;
        dim[i] = 2.0f * v.x * v.y * inv;
    }
}
__global__ void k_roll5(const float* lp, int n, float* dst) {
    size_t np = (size_t)n * n;
    for (size_t idx = blockIdx.x * (size_t)blockDim.x + threadIdx.x; idx < np;
         idx += (size_t)gridDim.x * blockDim.x) {
        int i = (int)(idx / n), j = (int)(idx - (size_t)i * n);
        dst[0 * np + idx] = lp[idx];
        dst[1 * np + idx] = lp[(size_t)i * n + ((j - 2 + n) & (n - 1))];
        dst[2 * np + idx] = lp[(size_t)i * n + ((j + 2) & (n - 1))];
        dst[3 * np + idx] = lp[(size_t)((i - 2 + n) & (n - 1)) * n + j];
        dst[4 * np + idx] = lp[(size_t)((i + 2) & (n - 1)) * n + j];
    }
}

// ---------------------------------------------------------------- spectral
__global__ void k_spec_hi0(const float2* in, float2* out, int n) {
    size_t np = (size_t)n * n;
    for (size_t idx = blockIdx.x * (size_t)blockDim.x + threadIdx.x; idx < np;
         idx += (size_t)gridDim.x * blockDim.x) {
        int k = (int)(idx / n), l = (int)(idx - (size_t)k * n);
        float lr, ang; ps_polar(k, l, n, 0, lr, ang);
        float g = ps_hi(lr, 0.0f);
        float2 z = in[idx];
        out[idx] = make_float2(z.x * g, z.y * g);
    }
}
__global__ void k_spec_lo0(const float2* in, float2* out, int n) {
    size_t np = (size_t)n * n;
    for (size_t idx = blockIdx.x * (size_t)blockDim.x + threadIdx.x; idx < np;
         idx += (size_t)gridDim.x * blockDim.x) {
        int k = (int)(idx / n), l = (int)(idx - (size_t)k * n);
        float lr, ang; ps_polar(k, l, n, 0, lr, ang);
        float g = ps_lo(lr, 0.0f);
        float2 z = in[idx];
        out[idx] = make_float2(z.x * g, z.y * g);
    }
}
// banddft = i * lodft * hi(lr, -(lev+1)) * am(b);   ((-1j)^3 == +i)
__global__ void k_spec_band(const float2* L, float2* out, int n, int lev, int b) {
    const float AMK = 1.7888543819998317f;  // 2*sqrt(0.8)
    size_t np = (size_t)n * n;
    for (size_t idx = blockIdx.x * (size_t)blockDim.x + threadIdx.x; idx < np;
         idx += (size_t)gridDim.x * blockDim.x) {
        int k = (int)(idx / n), l = (int)(idx - (size_t)k * n);
        float lr, ang; ps_polar(k, l, n, lev, lr, ang);
        float hm = ps_hi(lr, -(float)(lev + 1));
        float a = ang - 0.78539816339f * (float)b + PI_F;
        a = a - 6.28318530718f * floorf(a * 0.15915494309f);  // mod 2pi
        a -= PI_F;
        float c = cosf(a);
        float am = (fabsf(a) < 1.57079632679f) ? AMK * c * c * c : 0.0f;
        float g = hm * am;
        float2 z = L[idx];
        out[idx] = make_float2(-z.y * g, z.x * g);            // *i*g
    }
}
// crop (shifted-center window) then * lo(lr_new, -lev)
__global__ void k_spec_croplo(const float2* in, int nin, float2* out, int lev) {
    int nout = nin >> 1;
    int hi_ = nin >> 1, ho = nout >> 1, q = nin >> 2;
    size_t np = (size_t)nout * nout;
    for (size_t idx = blockIdx.x * (size_t)blockDim.x + threadIdx.x; idx < np;
         idx += (size_t)gridDim.x * blockDim.x) {
        int k2 = (int)(idx / nout), l2 = (int)(idx - (size_t)k2 * nout);
        int i2 = k2 ^ ho, j2 = l2 ^ ho;          // shifted (new)
        int ki = (i2 + q) ^ hi_, kj = (j2 + q) ^ hi_; // unshifted (old)
        float lr, ang; ps_polar(k2, l2, nout, lev, lr, ang);
        float g = ps_lo(lr, -(float)lev);
        float2 z = in[(size_t)ki * nin + kj];
        out[idx] = make_float2(z.x * g, z.y * g);
    }
}
// zero-pad spectrum n -> 2n (centered), unshifted in/out
__global__ void k_expand_pad(const float2* F, int n, float2* G) {
    int N = 2 * n, h = n >> 1;
    size_t np = (size_t)N * N;
    for (size_t idx = blockIdx.x * (size_t)blockDim.x + threadIdx.x; idx < np;
         idx += (size_t)gridDim.x * blockDim.x) {
        int k = (int)(idx / N), l = (int)(idx - (size_t)k * N);
        int i2 = k ^ n, j2 = l ^ n;              // shifted (2n grid)
        float2 v = make_float2(0.0f, 0.0f);
        if (i2 >= h && i2 < h + n && j2 >= h && j2 < h + n) {
            int ki = (i2 - h) ^ h, kj = (j2 - h) ^ h;
            v = F[(size_t)ki * n + kj];
        }
        G[idx] = v;
    }
}

// ---------------------------------------------------------------- reductions
__global__ void k_part_stats(const float* x, size_t n, float* part) {
    __shared__ float ss[256], sn[256], sx[256];
    float s = 0.0f, mn = FLT_BIG, mx = -FLT_BIG;
    for (size_t i = blockIdx.x * (size_t)blockDim.x + threadIdx.x; i < n;
         i += (size_t)gridDim.x * blockDim.x) {
        float v = x[i]; s += v; mn = fminf(mn, v); mx = fmaxf(mx, v);
    }
    int t = threadIdx.x;
    ss[t] = s; sn[t] = mn; sx[t] = mx; __syncthreads();
    for (int w = 128; w > 0; w >>= 1) {
        if (t < w) { ss[t] += ss[t + w]; sn[t] = fminf(sn[t], sn[t + w]);
                     sx[t] = fmaxf(sx[t], sx[t + w]); }
        __syncthreads();
    }
    if (t == 0) { part[blockIdx.x * 3 + 0] = ss[0];
                  part[blockIdx.x * 3 + 1] = sn[0];
                  part[blockIdx.x * 3 + 2] = sx[0]; }
}
__global__ void k_part_absum(const float* x, size_t n, float* part) {
    __shared__ float ss[256];
    float s = 0.0f;
    for (size_t i = blockIdx.x * (size_t)blockDim.x + threadIdx.x; i < n;
         i += (size_t)gridDim.x * blockDim.x) s += fabsf(x[i]);
    int t = threadIdx.x; ss[t] = s; __syncthreads();
    for (int w = 128; w > 0; w >>= 1) { if (t < w) ss[t] += ss[t + w]; __syncthreads(); }
    if (t == 0) part[blockIdx.x] = ss[0];
}
__global__ void k_part_cmom(const float* x, size_t n, const float* meanp,
                            float* part) {
    __shared__ float s2[256], s3[256], s4[256];
    float m = *meanp, a2 = 0.0f, a3 = 0.0f, a4 = 0.0f;
    for (size_t i = blockIdx.x * (size_t)blockDim.x + threadIdx.x; i < n;
         i += (size_t)gridDim.x * blockDim.x) {
        float c = x[i] - m, c2 = c * c;
        a2 += c2; a3 += c2 * c; a4 += c2 * c2;
    }
    int t = threadIdx.x; s2[t] = a2; s3[t] = a3; s4[t] = a4; __syncthreads();
    for (int w = 128; w > 0; w >>= 1) {
        if (t < w) { s2[t] += s2[t + w]; s3[t] += s3[t + w]; s4[t] += s4[t + w]; }
        __syncthreads();
    }
    if (t == 0) { part[blockIdx.x * 3 + 0] = s2[0];
                  part[blockIdx.x * 3 + 1] = s3[0];
                  part[blockIdx.x * 3 + 2] = s4[0]; }
}
__global__ void k_fin_scale(const float* part, int nb, int stride, int offs,
                            float scale, float* d1, float* d2) {
    __shared__ float ss[256];
    int t = threadIdx.x;
    ss[t] = (t < nb) ? part[offs + t * stride] : 0.0f; __syncthreads();
    for (int w = 128; w > 0; w >>= 1) { if (t < w) ss[t] += ss[t + w]; __syncthreads(); }
    if (t == 0) { float r = ss[0] * scale; if (d1) *d1 = r; if (d2) *d2 = r; }
}
__global__ void k_fin_minmax(const float* part, int nb, float* dmin, float* dmax) {
    __shared__ float sn[256], sx[256];
    int t = threadIdx.x;
    sn[t] = (t < nb) ? part[1 + t * 3] :  FLT_BIG;
    sx[t] = (t < nb) ? part[2 + t * 3] : -FLT_BIG;
    __syncthreads();
    for (int w = 128; w > 0; w >>= 1) {
        if (t < w) { sn[t] = fminf(sn[t], sn[t + w]); sx[t] = fmaxf(sx[t], sx[t + w]); }
        __syncthreads();
    }
    if (t == 0) { if (dmin) *dmin = sn[0]; if (dmax) *dmax = sx[0]; }
}
__global__ void k_fin_mom(const float* part, int nb, float invn, float* dvar,
                          float* dskew, float* dkurt, float* dstd) {
    __shared__ float s2[256], s3[256], s4[256];
    int t = threadIdx.x;
    s2[t] = (t < nb) ? part[t * 3 + 0] : 0.0f;
    s3[t] = (t < nb) ? part[t * 3 + 1] : 0.0f;
    s4[t] = (t < nb) ? part[t * 3 + 2] : 0.0f;
    __syncthreads();
    for (int w = 128; w > 0; w >>= 1) {
        if (t < w) { s2[t] += s2[t + w]; s3[t] += s3[t + w]; s4[t] += s4[t + w]; }
        __syncthreads();
    }
    if (t == 0) {
        float v = s2[0] * invn;
        if (dvar)  *dvar  = v;
        if (dskew) *dskew = s3[0] * invn / (v * sqrtf(v));
        if (dkurt) *dkurt = s4[0] * invn / (v * v);
        if (dstd)  *dstd  = sqrtf(v);
    }
}

// ---------------------------------------------------------------- autocorr
__global__ void k_ac_extract(const float2* ac, int n, float* dst, int stride) {
    int t = threadIdx.x;
    if (t >= 81) return;
    int di = t / 9, dj = t - di * 9, h = n >> 1;
    int ui = (h - 4 + di) ^ h, uj = (h - 4 + dj) ^ h;
    float ctr = ac[0].x;                        // shifted (h,h) == unshifted (0,0)
    dst[t * stride] = ac[(size_t)ui * n + uj].x / (ctr + 1e-12f);
}

// ---------------------------------------------------------------- WMMA grams
// C = S^T S for S (npix x 16), planes stored plane-major (16 x npix).
// 8-deep software pipeline: 8 independent global_load_b64 per lane, then
// 8 back-to-back v_wmma_f32_16x16x4_f32 accumulating into one v8f C.
__global__ void __launch_bounds__(32) k_gram16(const float* P, int npix,
                                               float* part) {
    int lane = threadIdx.x;
    int m = lane & 15, khi = lane >> 4;
    const float* row = P + (size_t)m * npix;
    v8f c = {0, 0, 0, 0, 0, 0, 0, 0};
    int wave = blockIdx.x, nw = gridDim.x;
#if __has_builtin(__builtin_amdgcn_wmma_f32_16x16x4_f32)
    const int step = nw * 32;                   // K advanced per wave per iter
    for (int k0 = wave * 32; k0 < npix; k0 += step) {
        int kk = k0 + 2 * khi;
        __builtin_prefetch(&row[kk + step], 0, 1);   // global_prefetch_b8
        v2f a[8];
#pragma unroll
        for (int u = 0; u < 8; ++u) {               // 8 x b64 load clause
            a[u][0] = row[kk + 4 * u];
            a[u][1] = row[kk + 4 * u + 1];
        }
#pragma unroll
        for (int u = 0; u < 8; ++u)                 // 8 x WMMA clause
            c = __builtin_amdgcn_wmma_f32_16x16x4_f32(false, a[u], false, a[u],
                                                      (short)0, c, false, false);
    }
#else
    for (int k0 = wave * 32; k0 < npix; k0 += nw * 32) {
        v16h a;
#pragma unroll
        for (int v = 0; v < 8; ++v)
#pragma unroll
            for (int hh = 0; hh < 2; ++hh) {
                int kk = (v < 4) ? (2 * v + hh + 8 * khi)
                                 : (16 + 2 * (v - 4) + hh + 8 * khi);
                a[2 * v + hh] = (_Float16)row[k0 + kk];
            }
        c = __builtin_amdgcn_wmma_f32_16x16x32_f16(false, a, false, a,
                                                   (short)0, c, false, false);
    }
#endif
    float* dst = part + (size_t)wave * 256;
#pragma unroll
    for (int r = 0; r < 8; ++r) dst[(r + 8 * khi) * 16 + m] = c[r];
}
__global__ void k_gram_reduce(const float* part, int nw, float* G) {
    int t = threadIdx.x;                        // 256 threads
    float s = 0.0f;
    for (int w = 0; w < nw; ++w) s += part[(size_t)w * 256 + t];
    G[t] = s;
}
__global__ void k_emit_normcc(const float* G, float invn, float* out, int base,
                              int rows, int cols, int rsel, int csel,
                              int sI, int sJ) {
    int t = threadIdx.x;
    if (t >= rows * cols) return;
    int i = t / cols, j = t - i * cols;
    float cij = G[(rsel + i) * 16 + (csel + j)] * invn;
    float sa = sqrtf(G[(rsel + i) * 17] * invn);
    float sb = sqrtf(G[(csel + j) * 17] * invn);
    out[base + i * sI + j * sJ] = cij / (sa * sb + 1e-12f);
}

// ---------------------------------------------------------------- host
extern "C" void kernel_launch(void* const* d_in, const int* in_sizes, int n_in,
                              void* d_out, int out_size, void* d_ws,
                              size_t ws_size, hipStream_t stream) {
    const float* img = (const float*)d_in[0];
    float* out = (float*)d_out;
    const size_t NP0 = 1024ull * 1024ull;

    // ---- carve workspace
    char* W = (char*)d_ws;
    size_t o = 0;
    auto carve = [&](size_t b) -> void* {
        void* p = W + o; o += (b + 255) & ~(size_t)255; return p;
    };
    float2* CW  = (float2*)carve(NP0 * sizeof(float2));
    float2* CW2 = (float2*)carve(NP0 * sizeof(float2));
    float2* L[5];
    for (int s = 0; s <= 4; ++s) {
        size_t n = (size_t)(1024 >> s);
        L[s] = (float2*)carve(n * n * sizeof(float2));
    }
    float2* BND[4][4];
    for (int s = 0; s < 4; ++s) {
        size_t n = (size_t)(1024 >> s);
        for (int b = 0; b < 4; ++b)
            BND[s][b] = (float2*)carve(n * n * sizeof(float2));
    }
    float* G1   = (float*)carve(16 * NP0 * sizeof(float));
    float* G2   = (float*)carve(16 * NP0 * sizeof(float));
    float* PART = (float*)carve(RB * 4 * sizeof(float));
    float* GP   = (float*)carve((size_t)GW * 256 * sizeof(float));
    float* GA   = (float*)carve(256 * sizeof(float));
    float* GB   = (float*)carve(256 * sizeof(float));
    float* SC   = (float*)carve(64 * sizeof(float));
    (void)in_sizes; (void)n_in; (void)out_size; (void)ws_size;

    auto gs = [](size_t n) -> dim3 {
        size_t b = (n + 255) / 256; if (b > 8192) b = 8192;
        return dim3((unsigned)b);
    };
    auto fft2 = [&](float2* p, int n, int inv) {
        int logn = 0; while ((1 << logn) < n) ++logn;
        k_fft_lines<<<dim3(n), dim3(256), 0, stream>>>(p, n, logn, 1, n, inv);
        k_fft_lines<<<dim3(n), dim3(256), 0, stream>>>(p, n, logn, n, 1, inv);
    };
    auto reduce_mean = [&](const float* x, size_t n, float* d1, float* d2) {
        k_part_stats<<<dim3(RB), dim3(256), 0, stream>>>(x, n, PART);
        k_fin_scale<<<dim3(1), dim3(256), 0, stream>>>(PART, RB, 3, 0,
                                                       1.0f / (float)n, d1, d2);
    };
    auto moments = [&](const float* x, size_t n, const float* meanp, float* dv,
                       float* dsk, float* dku, float* dst) {
        k_part_cmom<<<dim3(RB), dim3(256), 0, stream>>>(x, n, meanp, PART);
        k_fin_mom<<<dim3(1), dim3(256), 0, stream>>>(PART, RB, 1.0f / (float)n,
                                                     dv, dsk, dku, dst);
    };
    auto autocorr = [&](const float* plane, int n, float2* work, float* dst,
                        int stride) {
        size_t np = (size_t)n * n;
        reduce_mean(plane, np, SC + 1, (float*)nullptr);
        k_r2c<<<gs(np), dim3(256), 0, stream>>>(plane, work, np, SC + 1);
        fft2(work, n, 0);
        k_power<<<gs(np), dim3(256), 0, stream>>>(work, np);
        fft2(work, n, 1);
        k_ac_extract<<<dim3(1), dim3(128), 0, stream>>>(work, n, dst, stride);
    };

    // ---- zero output (com/cor/csm/csr have structurally-zero regions)
    k_memset_f32<<<gs(2461), dim3(256), 0, stream>>>(out, 0.0f, 2461);

    // ---- pixel stats
    k_part_stats<<<dim3(RB), dim3(256), 0, stream>>>(img, NP0, PART);
    k_fin_scale<<<dim3(1), dim3(256), 0, stream>>>(PART, RB, 3, 0,
                                                   1.0f / (float)NP0, SC + 0, out + 0);
    k_fin_minmax<<<dim3(1), dim3(256), 0, stream>>>(PART, RB, out + 4, out + 5);
    moments(img, NP0, SC + 0, out + 1, out + 2, out + 3, (float*)nullptr);

    // ---- image DFT (unshifted; masks use XOR-shift coords)
    k_r2c<<<gs(NP0), dim3(256), 0, stream>>>(img, CW, NP0, (const float*)nullptr);
    fft2(CW, 1024, 0);

    // ---- hipass: stats + |.|-mean + var_hp
    k_spec_hi0<<<gs(NP0), dim3(256), 0, stream>>>(CW, CW2, 1024);
    fft2(CW2, 1024, 1);
    k_c2real<<<gs(NP0), dim3(256), 0, stream>>>(CW2, G1, NP0);
    k_part_absum<<<dim3(RB), dim3(256), 0, stream>>>(G1, NP0, PART);
    k_fin_scale<<<dim3(1), dim3(256), 0, stream>>>(PART, RB, 1, 0,
                                                   1.0f / (float)NP0, out + 6,
                                                   (float*)nullptr);
    reduce_mean(G1, NP0, SC + 0, (float*)nullptr);
    moments(G1, NP0, SC + 0, out + 2460, (float*)nullptr, (float*)nullptr,
            (float*)nullptr);

    // ---- lowpass chain L0..L4
    k_spec_lo0<<<gs(NP0), dim3(256), 0, stream>>>(CW, L[0], 1024);
    for (int s = 0; s < 4; ++s) {
        int n = 1024 >> s; size_t np2 = (size_t)(n >> 1) * (n >> 1);
        k_spec_croplo<<<gs(np2), dim3(256), 0, stream>>>(L[s], n, L[s + 1], s + 1);
    }

    // ---- oriented bands (spatial complex)
    for (int s = 0; s < 4; ++s) {
        int n = 1024 >> s; size_t np = (size_t)n * n;
        for (int b = 0; b < 4; ++b) {
            k_spec_band<<<gs(np), dim3(256), 0, stream>>>(L[s], BND[s][b], n, s, b);
            fft2(BND[s][b], n, 1);
        }
    }

    // ---- per-scale statistics
    for (int s = 0; s < 4; ++s) {
        int n = 1024 >> s; size_t np = (size_t)n * n;
        float invn = 1.0f / (float)np;
        k_memset_f32<<<gs(16 * np), dim3(256), 0, stream>>>(G1, 0.0f, 16 * np);
        k_memset_f32<<<gs(16 * np), dim3(256), 0, stream>>>(G2, 0.0f, 16 * np);
        for (int b = 0; b < 4; ++b) {
            float* magp = G1 + (size_t)b * np;
            float* rep  = G2 + (size_t)b * np;
            k_band_to_planes<<<gs(np), dim3(256), 0, stream>>>(BND[s][b], np,
                                                               magp, rep);
            reduce_mean(magp, np, SC + 0, out + 7 + 4 * s + b);  // mag_means
            k_center<<<gs(np), dim3(256), 0, stream>>>(magp, np, SC + 0);
            reduce_mean(rep, np, SC + 0, (float*)nullptr);
            k_center<<<gs(np), dim3(256), 0, stream>>>(rep, np, SC + 0);
            autocorr(magp, n, CW, out + 24 + 4 * s + b, 16);     // acm
        }
        if (s < 3) {
            int n2 = n >> 1; size_t np2 = (size_t)n2 * n2;
            for (int b = 0; b < 4; ++b) {
                (void)hipMemcpyAsync(CW, BND[s + 1][b], np2 * sizeof(float2),
                                     hipMemcpyDeviceToDevice, stream);
                fft2(CW, n2, 0);
                k_expand_pad<<<gs(np), dim3(256), 0, stream>>>(CW, n2, CW2);
                fft2(CW2, n, 1);
                k_cscale<<<gs(np), dim3(256), 0, stream>>>(CW2, np, 4.0f);
                k_dp_planes<<<gs(np), dim3(256), 0, stream>>>(
                    CW2, np, G1 + (size_t)(4 + b) * np,
                    G2 + (size_t)(4 + b) * np, G2 + (size_t)(8 + b) * np);
            }
            for (int p = 4; p < 8; ++p) {
                reduce_mean(G1 + (size_t)p * np, np, SC + 0, (float*)nullptr);
                k_center<<<gs(np), dim3(256), 0, stream>>>(G1 + (size_t)p * np,
                                                           np, SC + 0);
            }
            for (int p = 4; p < 12; ++p) {
                reduce_mean(G2 + (size_t)p * np, np, SC + 0, (float*)nullptr);
                k_center<<<gs(np), dim3(256), 0, stream>>>(G2 + (size_t)p * np,
                                                           np, SC + 0);
            }
        } else {
            // lp = expand2(lowpass).real, centered; 5 rolled planes
            k_expand_pad<<<gs(np), dim3(256), 0, stream>>>(L[4], 64, CW2);
            fft2(CW2, 128, 1);
            k_cscale<<<gs(np), dim3(256), 0, stream>>>(CW2, np, 4.0f);
            float* lp = G1 + 4 * np;
            k_c2real<<<gs(np), dim3(256), 0, stream>>>(CW2, lp, np);
            reduce_mean(lp, np, SC + 0, (float*)nullptr);
            k_center<<<gs(np), dim3(256), 0, stream>>>(lp, np, SC + 0);
            k_roll5<<<gs(np), dim3(256), 0, stream>>>(lp, n, G2 + 4 * np);
        }
        // grams via WMMA
        k_gram16<<<dim3(GW), dim3(32), 0, stream>>>(G1, (int)np, GP);
        k_gram_reduce<<<dim3(1), dim3(256), 0, stream>>>(GP, GW, GA);
        k_gram16<<<dim3(GW), dim3(32), 0, stream>>>(G2, (int)np, GP);
        k_gram_reduce<<<dim3(1), dim3(256), 0, stream>>>(GP, GW, GB);
        // com (4,4,5), cor (8,8,5), csm (4,4,4), csr (8,8,4)
        k_emit_normcc<<<dim3(1), dim3(64), 0, stream>>>(GA, invn, out, 1740 + s,
                                                        4, 4, 0, 0, 20, 5);
        k_emit_normcc<<<dim3(1), dim3(64), 0, stream>>>(GB, invn, out, 1884 + s,
                                                        4, 4, 0, 0, 40, 5);
        if (s < 3) {
            k_emit_normcc<<<dim3(1), dim3(64), 0, stream>>>(GA, invn, out,
                                                            1820 + s, 4, 4, 0, 4,
                                                            16, 4);
            k_emit_normcc<<<dim3(1), dim3(64), 0, stream>>>(GB, invn, out,
                                                            2204 + s, 4, 8, 0, 4,
                                                            32, 4);
        } else {
            k_emit_normcc<<<dim3(1), dim3(64), 0, stream>>>(GB, invn, out,
                                                            2204 + 3, 4, 5, 0, 4,
                                                            32, 4);
            k_emit_normcc<<<dim3(1), dim3(64), 0, stream>>>(GB, invn, out,
                                                            1884 + 4, 5, 5, 4, 4,
                                                            40, 5);
        }
    }

    // ---- reconstructions: skew/kurt/std + acr + |lowpass| mean
    for (int s = 0; s <= 4; ++s) {
        int n = 1024 >> s; size_t np = (size_t)n * n;
        (void)hipMemcpyAsync(CW, L[s], np * sizeof(float2),
                             hipMemcpyDeviceToDevice, stream);
        fft2(CW, n, 1);
        float* rec = G1;
        k_c2real<<<gs(np), dim3(256), 0, stream>>>(CW, rec, np);
        reduce_mean(rec, np, SC + 0, (float*)nullptr);
        moments(rec, np, SC + 0, (float*)nullptr, out + 1320 + s,
                out + 1325 + s, out + 1735 + s);
        autocorr(rec, n, CW2, out + 1330 + s, 5);
        if (s == 4) {
            k_part_absum<<<dim3(RB), dim3(256), 0, stream>>>(rec, np, PART);
            k_fin_scale<<<dim3(1), dim3(256), 0, stream>>>(PART, RB, 1, 0,
                                                           1.0f / (float)np,
                                                           out + 23,
                                                           (float*)nullptr);
        }
    }
}
